// LSTMModel_69226282877089
// MI455X (gfx1250) — matrix operand order repaired and checked by reading, not compiled, not run on hardware
//
#include <hip/hip_runtime.h>
#include <hip/hip_bf16.h>
#include <stdint.h>

// Fixed problem sizes from the reference
#define B_  256
#define T_  512
#define D_  512
#define H_  512
#define G4_ 2048            // 4*H
#define KP_ 256             // K/2 packed-pair count (D and H are both 512)
#define HDW 65536           // B_*H_/2 : dwords per bf16 [256][512] buffer

#define NBLK  64
#define BLOCK 128           // 4 waves per block (wave32)

typedef __attribute__((ext_vector_type(16))) __bf16 v16bf;
typedef __attribute__((ext_vector_type(8)))  float  v8f;
typedef unsigned int u32x4 __attribute__((ext_vector_type(4)));
typedef float        f32x4 __attribute__((ext_vector_type(4)));

union Frag {
  unsigned u[8];
  u32x4    q[2];
  v16bf    v;
};

__device__ __forceinline__ unsigned short f2bf(float f) {
  unsigned x = __float_as_uint(f);
  unsigned r = x + 0x7fffu + ((x >> 16) & 1u);   // round-to-nearest-even
  return (unsigned short)(r >> 16);
}
__device__ __forceinline__ unsigned pk2(float lo, float hi) {
  return (unsigned)f2bf(lo) | ((unsigned)f2bf(hi) << 16);
}
__device__ __forceinline__ float sigm_(float x) { return 1.0f / (1.0f + __expf(-x)); }
__device__ __forceinline__ float tanh_(float x) { return 2.0f / (1.0f + __expf(-2.0f * x)) - 1.0f; }

// Monotonic-counter grid barrier: target = generation * NBLK.
// Counter zeroed by hipMemsetAsync each launch.
__device__ __forceinline__ void grid_barrier(unsigned* ctr, unsigned target) {
  __threadfence();
  __syncthreads();
  if (threadIdx.x == 0) {
    __hip_atomic_fetch_add(ctr, 1u, __ATOMIC_RELEASE, __HIP_MEMORY_SCOPE_AGENT);
    while (__hip_atomic_load(ctr, __ATOMIC_ACQUIRE, __HIP_MEMORY_SCOPE_AGENT) < target) {
      __builtin_amdgcn_s_sleep(1);
    }
  }
  __syncthreads();
  __threadfence();
}

// Convert one time-slice x[:, t, :] (fp32) into packed bf16 row-major buffer.
// 65536 dwords over 8192 threads -> 8 dwords (16 floats) per thread.
__device__ __forceinline__ void convert_x_slice(const float* __restrict__ x, int t,
                                                unsigned* __restrict__ xdst, int tid) {
  const int d0  = tid * 8;            // first dword this thread produces
  const int row = d0 >> 8;            // 256 dwords per row
  const int kp0 = d0 & 255;
  const float* px = x + ((size_t)row * T_ + t) * D_ + 2 * kp0;
  f32x4 f0 = *(const f32x4*)(px);
  f32x4 f1 = *(const f32x4*)(px + 4);
  f32x4 f2 = *(const f32x4*)(px + 8);
  f32x4 f3 = *(const f32x4*)(px + 12);
  u32x4 o0, o1;
  o0.x = pk2(f0.x, f0.y); o0.y = pk2(f0.z, f0.w);
  o0.z = pk2(f1.x, f1.y); o0.w = pk2(f1.z, f1.w);
  o1.x = pk2(f2.x, f2.y); o1.y = pk2(f2.z, f2.w);
  o1.z = pk2(f3.x, f3.y); o1.w = pk2(f3.z, f3.w);
  *(u32x4*)(xdst + d0)     = o0;
  *(u32x4*)(xdst + d0 + 4) = o1;
}

__global__ __launch_bounds__(BLOCK, 1)
void lstm_wmma_kernel(const float* __restrict__ x,      // [B,T,D]
                      const float* __restrict__ Wx,     // [D,4H]
                      const float* __restrict__ Wh,     // [H,4H]
                      const float* __restrict__ bias,   // [4H]
                      const float* __restrict__ Wfc,    // [H]
                      const float* __restrict__ bfc,    // [1]
                      float* __restrict__ out,          // [B]
                      unsigned* __restrict__ ctr,
                      unsigned* __restrict__ Wxp,       // packed bf16 pairs [2048][256]
                      unsigned* __restrict__ Whp,       // packed bf16 pairs [2048][256]
                      unsigned* __restrict__ hbf,       // 2 ping-pong bf16 h bufs (dwords)
                      unsigned* __restrict__ xbf,       // 2 ping-pong bf16 x_t bufs (dwords)
                      float* __restrict__ hf,           // fp32 h (for final fc)
                      float* __restrict__ cst)          // fp32 c state
{
  const int tid = (int)(blockIdx.x * BLOCK + threadIdx.x);
  const int nth = NBLK * BLOCK;

  // ---------- setup: pack weights; convert x_0; zero state ----------
  // Wp[col][kp] = pack_bf16(W[2kp][col], W[2kp+1][col])  (pairs along K)
  for (int i = tid; i < G4_ * KP_; i += nth) {
    const int col = i >> 8;
    const int kp  = i & (KP_ - 1);
    Wxp[(size_t)col * KP_ + kp] =
        pk2(Wx[(size_t)(2 * kp) * G4_ + col], Wx[(size_t)(2 * kp + 1) * G4_ + col]);
    Whp[(size_t)col * KP_ + kp] =
        pk2(Wh[(size_t)(2 * kp) * G4_ + col], Wh[(size_t)(2 * kp + 1) * G4_ + col]);
  }
  for (int i = tid; i < HDW; i += nth) hbf[i] = 0u;       // h buffer 0 = 0
  for (int i = tid; i < B_ * H_; i += nth) cst[i] = 0.0f; // c = 0
  convert_x_slice(x, 0, xbf, tid);                        // x buffer 0 = bf16(x[:,0,:])
  grid_barrier(ctr, NBLK);

  // Wave ownership: 256 waves = 8 row groups (32 rows) x 32 col tiles (16 cols)
  const int wave = (int)((blockIdx.x << 2) | (threadIdx.x >> 5));
  const int lane = (int)(threadIdx.x & 31);
  const int half = lane >> 4;
  const int ln   = lane & 15;
  const int rowBase = (wave >> 5) << 5;  // batch rows [rowBase, rowBase+32)
  const int n0      = (wave & 31) << 4;  // hidden cols [n0, n0+16)

  float biasv[4];
#pragma unroll
  for (int g = 0; g < 4; ++g) biasv[g] = bias[g * H_ + n0 + ln];

  // Per-lane base offsets (constant across steps)
  const size_t wOff  = (size_t)(n0 + ln) * KP_ + half * 8;      // + g*H_*KP_ + ks*16
  const int    aOff0 = (rowBase + ln) * (H_ / 2) + half * 8;    // rt adds 16*(H_/2)

  unsigned cur = 0;
#pragma unroll 1
  for (int t = 0; t < T_; ++t) {
    v8f acc[8];                      // [rt][gate], init with bias
#pragma unroll
    for (int i = 0; i < 8; ++i) {
      const float bv = biasv[i & 3];
#pragma unroll
      for (int r = 0; r < 8; ++r) acc[i][r] = bv;
    }

    const unsigned* __restrict__ hsrc = hbf + (size_t)cur * HDW;
    const unsigned* __restrict__ xsrc = xbf + (size_t)cur * HDW;
    unsigned* __restrict__ hdst = hbf + (size_t)(cur ^ 1u) * HDW;
    unsigned* __restrict__ xdst = xbf + (size_t)(cur ^ 1u) * HDW;

    // ---- two GEMM phases, identical layout: gates += A @ W ----
#pragma unroll
    for (int ph = 0; ph < 2; ++ph) {
      const unsigned* __restrict__ Wp  = ph ? Whp : Wxp;
      const unsigned* __restrict__ src = ph ? hsrc : xsrc;
#pragma unroll 2
      for (int ks = 0; ks < 16; ++ks) {
        Frag bw[4];
#pragma unroll
        for (int g = 0; g < 4; ++g) {
          const unsigned* p = Wp + (size_t)g * H_ * KP_ + wOff + ks * 16;
          bw[g].q[0] = *(const u32x4*)(p);
          bw[g].q[1] = *(const u32x4*)(p + 4);
        }
#pragma unroll
        for (int rt = 0; rt < 2; ++rt) {
          const unsigned* pa = src + aOff0 + rt * 16 * (H_ / 2) + ks * 16;
          Frag a;
          a.q[0] = *(const u32x4*)(pa);
          a.q[1] = *(const u32x4*)(pa + 4);
#pragma unroll
          for (int g = 0; g < 4; ++g)
            acc[rt * 4 + g] = __builtin_amdgcn_wmma_f32_16x16x32_bf16(
                false, a.v, false, bw[g].v, (short)0, acc[rt * 4 + g], false, false);
        }
      }
    }

    // ---- LSTM cell (fp32): this wave owns rows [rowBase,+32), cols [n0,+16) ----
    unsigned short* hdst16 = (unsigned short*)hdst;
#pragma unroll
    for (int rt = 0; rt < 2; ++rt) {
#pragma unroll
      for (int r = 0; r < 8; ++r) {
        const int row = rowBase + rt * 16 + half * 8 + r;   // C/D layout: M = r + half*8
        const int col = n0 + ln;                            // N = lane%16
        const size_t idx = (size_t)row * H_ + col;
        const float iv = acc[rt * 4 + 0][r];
        const float fv = acc[rt * 4 + 1][r];
        const float gv = acc[rt * 4 + 2][r];
        const float ov = acc[rt * 4 + 3][r];
        const float cold = cst[idx];
        const float cn = sigm_(fv) * cold + sigm_(iv) * tanh_(gv);
        const float hn = sigm_(ov) * tanh_(cn);
        cst[idx]    = cn;
        hf[idx]     = hn;
        hdst16[idx] = f2bf(hn);
      }
    }

    // ---- pipeline: convert next x slice for step t+1 (once, cooperatively) ----
    if (t + 1 < T_) convert_x_slice(x, t + 1, xdst, tid);

    cur ^= 1u;
    grid_barrier(ctr, (unsigned)(t + 2) * NBLK);
  }

  // ---- final fc: out[b] = h_T[b,:] @ Wfc + bfc ----
  if (blockIdx.x == 0) {
    for (int row = (int)threadIdx.x; row < B_; row += BLOCK) {
      float s = bfc[0];
      const float* hr = hf + (size_t)row * H_;
#pragma unroll 8
      for (int k = 0; k < H_; ++k) s += hr[k] * Wfc[k];
      out[row] = s;
    }
  }
}

extern "C" void kernel_launch(void* const* d_in, const int* in_sizes, int n_in,
                              void* d_out, int out_size, void* d_ws, size_t ws_size,
                              hipStream_t stream) {
  const float* x    = (const float*)d_in[0];
  const float* Wx   = (const float*)d_in[1];
  const float* Wh   = (const float*)d_in[2];
  const float* bias = (const float*)d_in[3];
  const float* Wfc  = (const float*)d_in[4];
  const float* bfc  = (const float*)d_in[5];
  float* out = (float*)d_out;

  // Workspace layout (all 256B-aligned):
  //   [0,256)     barrier counter
  //   +2MB        Wx packed bf16 pairs
  //   +2MB        Wh packed bf16 pairs
  //   +512KB      h bf16 ping-pong (2 x 256x512)
  //   +512KB      x_t bf16 ping-pong (2 x 256x512)
  //   +512KB      h fp32
  //   +512KB      c fp32
  uint8_t* ws = (uint8_t*)d_ws;
  const size_t WSZ = (size_t)G4_ * KP_ * 4;   // 2 MB per packed weight
  const size_t HBF = 2u * (size_t)HDW * 4;    // 512 KB per ping-pong pair
  unsigned* ctr = (unsigned*)ws;
  unsigned* Wxp = (unsigned*)(ws + 256);
  unsigned* Whp = (unsigned*)(ws + 256 + WSZ);
  unsigned* hbf = (unsigned*)(ws + 256 + 2 * WSZ);
  unsigned* xbf = (unsigned*)(ws + 256 + 2 * WSZ + HBF);
  float*    hf  = (float*)(ws + 256 + 2 * WSZ + 2 * HBF);
  float*    cst = (float*)(ws + 256 + 2 * WSZ + 2 * HBF + (size_t)B_ * H_ * 4);

  hipMemsetAsync(ctr, 0, 256, stream);  // graph-capture legal
  lstm_wmma_kernel<<<NBLK, BLOCK, 0, stream>>>(x, Wx, Wh, bias, Wfc, bfc, out,
                                               ctr, Wxp, Whp, hbf, xbf, hf, cst);
}
